// MambaScan1D_69638599737481
// MI455X (gfx1250) — matrix operand order repaired and checked by reading, not compiled, not run on hardware
//
#include <hip/hip_runtime.h>
#include <hip/hip_bf16.h>

typedef __bf16 bf16;
typedef __attribute__((ext_vector_type(16))) __bf16 v16bf;
typedef __attribute__((ext_vector_type(8)))  float  v8f;
typedef __attribute__((__vector_size__(4 * sizeof(int)))) int v4i;

#define WG 256

#if defined(__has_builtin)
#  if __has_builtin(__builtin_amdgcn_global_load_async_to_lds_b128) && \
      __has_builtin(__builtin_amdgcn_s_wait_asynccnt)
#    define HAVE_ASYNC_LDS 1
#  endif
#endif
#ifndef HAVE_ASYNC_LDS
#  define HAVE_ASYNC_LDS 0
#endif

#if HAVE_ASYNC_LDS
#define GLB_V4I(p) ((__attribute__((address_space(1))) v4i*)(p))
#define LDS_V4I(p) ((__attribute__((address_space(3))) v4i*)(p))
#define ASYNC_CP_B128(dst_lds, src_glb) \
    __builtin_amdgcn_global_load_async_to_lds_b128(GLB_V4I(src_glb), LDS_V4I(dst_lds), 0, 0)
#endif

__device__ __forceinline__ float sigmoid_fast(float v) {
    return 1.0f / (1.0f + __expf(-v));
}

// ---------------- generic f32 -> bf16 convert ----------------
__global__ void __launch_bounds__(WG) k_cvt_bf16(const float* __restrict__ in,
                                                 bf16* __restrict__ out, int n) {
    for (int i = blockIdx.x * WG + threadIdx.x; i < n; i += gridDim.x * WG)
        out[i] = (bf16)in[i];
}

// ---------------- LayerNorm (one row per block), writes bf16 ----------------
__global__ void __launch_bounds__(WG) k_layernorm(const float* __restrict__ x,
                                                  const float* __restrict__ g,
                                                  const float* __restrict__ b,
                                                  bf16* __restrict__ out, int D) {
    __shared__ float red[WG];
    int row = blockIdx.x;
    const float* xr = x + (size_t)row * D;
    float s = 0.f;
    for (int i = threadIdx.x; i < D; i += WG) s += xr[i];
    red[threadIdx.x] = s; __syncthreads();
    for (int o = WG / 2; o > 0; o >>= 1) {
        if (threadIdx.x < o) red[threadIdx.x] += red[threadIdx.x + o];
        __syncthreads();
    }
    float mu = red[0] / D;
    __syncthreads();
    float v = 0.f;
    for (int i = threadIdx.x; i < D; i += WG) { float d0 = xr[i] - mu; v += d0 * d0; }
    red[threadIdx.x] = v; __syncthreads();
    for (int o = WG / 2; o > 0; o >>= 1) {
        if (threadIdx.x < o) red[threadIdx.x] += red[threadIdx.x + o];
        __syncthreads();
    }
    float inv = rsqrtf(red[0] / D + 1e-5f);
    bf16* orow = out + (size_t)row * D;
    for (int i = threadIdx.x; i < D; i += WG)
        orow[i] = (bf16)((xr[i] - mu) * inv * g[i] + b[i]);
}

// ---------------- bf16 WMMA GEMM: C[M,N] = A[M,K] * B[N,K]^T (+bias_n) (+residual) ----
// Block tile 128x64, K-step 64, double-buffered LDS, ONE barrier per K-step, with the
// async global->LDS copy of stage s+1 issued right after the barrier so it overlaps the
// 8 WMMAs of stage s. 8 waves (4x2); each wave owns a 32x32 patch.
__global__ void __launch_bounds__(WG) k_gemm_bf16(const bf16* __restrict__ A,
                                                  const bf16* __restrict__ B,
                                                  float* __restrict__ C,
                                                  const float* __restrict__ bias,
                                                  const float* __restrict__ residual,
                                                  int M, int N, int K) {
    __shared__ __align__(16) bf16 As[2][128][72];   // 64 cols + 8 pad (144B rows, 16B-mult)
    __shared__ __align__(16) bf16 Bs[2][64][72];
    const int tid  = threadIdx.x;
    const int wave = tid >> 5, lane = tid & 31;
    const int wm = wave >> 1, wn = wave & 1;
    const int half = lane >> 4, r = lane & 15;
    const int m0 = blockIdx.y * 128;
    const int n0 = blockIdx.x * 64;

    // tile-load mapping: 16B (8 x bf16) chunks; 8 chunks per row of 64 cols
    const int lr   = tid >> 3;            // 0..31
    const int lcol = (tid & 7) * 8;       // 0..56

    const bf16* gA[4];
    const bf16* gB[2];
    #pragma unroll
    for (int q = 0; q < 4; ++q) gA[q] = A + (size_t)(m0 + lr + 32 * q) * K + lcol;
    #pragma unroll
    for (int q = 0; q < 2; ++q) gB[q] = B + (size_t)(n0 + lr + 32 * q) * K + lcol;

    v8f acc[2][2] = {};

    // ---- preload stage 0 (k0 = 0)
#if HAVE_ASYNC_LDS
    #pragma unroll
    for (int q = 0; q < 4; ++q) ASYNC_CP_B128(&As[0][lr + 32 * q][lcol], gA[q]);
    #pragma unroll
    for (int q = 0; q < 2; ++q) ASYNC_CP_B128(&Bs[0][lr + 32 * q][lcol], gB[q]);
#else
    #pragma unroll
    for (int q = 0; q < 4; ++q) *(float4*)&As[0][lr + 32 * q][lcol] = *(const float4*)gA[q];
    #pragma unroll
    for (int q = 0; q < 2; ++q) *(float4*)&Bs[0][lr + 32 * q][lcol] = *(const float4*)gB[q];
#endif

    int stage = 0;
    for (int k0 = 0; k0 < K; k0 += 64) {
        // Barrier FIRST: the only pending async ops here are the current stage's
        // (issued one compute-phase ago) -> the implicit asynccnt/dscnt wait is exact.
        __syncthreads();

        const int  nxt  = stage ^ 1;
        const bool more = (k0 + 64) < K;

        // ---- issue next-stage tile fetch AFTER the barrier (overlaps the WMMAs below)
#if HAVE_ASYNC_LDS
        if (more) {
            #pragma unroll
            for (int q = 0; q < 4; ++q)
                ASYNC_CP_B128(&As[nxt][lr + 32 * q][lcol], gA[q] + k0 + 64);
            #pragma unroll
            for (int q = 0; q < 2; ++q)
                ASYNC_CP_B128(&Bs[nxt][lr + 32 * q][lcol], gB[q] + k0 + 64);
        }
#else
        float4 ta[4], tb[2];
        if (more) {
            #pragma unroll
            for (int q = 0; q < 4; ++q) ta[q] = *(const float4*)(gA[q] + k0 + 64);
            #pragma unroll
            for (int q = 0; q < 2; ++q) tb[q] = *(const float4*)(gB[q] + k0 + 64);
        }
#endif

        // ---- two 32-K chunks from current stage: 8 WMMAs per wave per iteration
        #pragma unroll
        for (int kk = 0; kk < 2; ++kk) {
            const int kb0 = kk * 32;
            v16bf af[2], bfr[2];
            #pragma unroll
            for (int j = 0; j < 16; ++j) {
                // 16-bit A 16x32: lanes<16 hold K {0..7,16..23}, lanes>=16 hold {8..15,24..31}
                int ka = kb0 + ((j < 8) ? j : j + 8) + half * 8;
                // 16-bit B 32x16: lanes<16 hold K {0..7,16..23}, lanes>=16 hold {8..15,24..31}
                int kb = kb0 + ((j < 8) ? j : j + 8) + half * 8;
                af[0][j]  = As[stage][wm * 32 + r][ka];
                af[1][j]  = As[stage][wm * 32 + 16 + r][ka];
                bfr[0][j] = Bs[stage][wn * 32 + r][kb];
                bfr[1][j] = Bs[stage][wn * 32 + 16 + r][kb];
            }
            #pragma unroll
            for (int i = 0; i < 2; ++i)
                #pragma unroll
                for (int j2 = 0; j2 < 2; ++j2)
                    acc[i][j2] = __builtin_amdgcn_wmma_f32_16x16x32_bf16(
                        false, af[i], false, bfr[j2], (short)0, acc[i][j2], false, false);
        }

#if !HAVE_ASYNC_LDS
        if (more) {
            #pragma unroll
            for (int q = 0; q < 4; ++q) *(float4*)&As[nxt][lr + 32 * q][lcol] = ta[q];
            #pragma unroll
            for (int q = 0; q < 2; ++q) *(float4*)&Bs[nxt][lr + 32 * q][lcol] = tb[q];
        }
#endif
        stage = nxt;
    }

    // C/D layout: VGPR g -> M = g + half*8, N = r (within each 16x16 tile)
    #pragma unroll
    for (int i = 0; i < 2; ++i)
        #pragma unroll
        for (int j2 = 0; j2 < 2; ++j2)
            #pragma unroll
            for (int g = 0; g < 8; ++g) {
                int row = m0 + wm * 32 + i * 16 + g + half * 8;
                int col = n0 + wn * 32 + j2 * 16 + r;
                float v = acc[i][j2][g];
                if (bias) v += bias[col];
                size_t ix = (size_t)row * N + col;
                if (residual) v += residual[ix];
                C[ix] = v;
            }
}

// ---------------- depthwise conv3 along L + per-channel stat accumulation ----------
#define CONV_TPB 64
__global__ void __launch_bounds__(WG) k_dwconv(const float* __restrict__ xp,
                                               const float* __restrict__ cw,
                                               const float* __restrict__ cb,
                                               float* __restrict__ out,
                                               float* __restrict__ sums,
                                               float* __restrict__ sumsq,
                                               int B, int L, int D, int NC, int coff) {
    int d = blockIdx.x * WG + threadIdx.x;       // channel
    int chunks_per_b = L / CONV_TPB;
    int b  = blockIdx.y / chunks_per_b;
    int t0 = (blockIdx.y % chunks_per_b) * CONV_TPB;
    float w0 = cw[d * 3 + 0], w1 = cw[d * 3 + 1], w2 = cw[d * 3 + 2], bb = cb[d];
    const float* base = xp + (size_t)b * L * NC + coff + d;
    float* ob = out + (size_t)b * L * D + d;
    float s = 0.f, s2 = 0.f;
    for (int tt = 0; tt < CONV_TPB; ++tt) {
        int t = t0 + tt;
        float xm = (t > 0)     ? base[(size_t)(t - 1) * NC] : 0.f;
        float x0 = base[(size_t)t * NC];
        float xq = (t < L - 1) ? base[(size_t)(t + 1) * NC] : 0.f;
        float o = w0 * xm + w1 * x0 + w2 * xq + bb;
        ob[(size_t)t * D] = o;
        s += o; s2 += o * o;
    }
    atomicAdd(&sums[d], s);
    atomicAdd(&sumsq[d], s2);
}

__global__ void __launch_bounds__(WG) k_bn_final(const float* __restrict__ sums,
                                                 const float* __restrict__ sumsq,
                                                 const float* __restrict__ g,
                                                 const float* __restrict__ b,
                                                 float* __restrict__ scale,
                                                 float* __restrict__ shift,
                                                 int D, float invBL) {
    int d = blockIdx.x * WG + threadIdx.x;
    if (d >= D) return;
    float mu  = sums[d] * invBL;
    float var = sumsq[d] * invBL - mu * mu;
    float sc  = g[d] * rsqrtf(var + 1e-5f);
    scale[d] = sc;
    shift[d] = b[d] - mu * sc;
}

// ---------------- apply BN + SiLU in-place (f32) and emit bf16 copy ----------------
__global__ void __launch_bounds__(WG) k_bn_silu(float* __restrict__ io,
                                                bf16* __restrict__ ubf,
                                                const float* __restrict__ scale,
                                                const float* __restrict__ shift,
                                                int dmask, int n) {
    for (int i = blockIdx.x * WG + threadIdx.x; i < n; i += gridDim.x * WG) {
        int d = i & dmask;
        float v = io[i] * scale[d] + shift[d];
        float u = v * sigmoid_fast(v);
        io[i]  = u;
        ubf[i] = (bf16)u;
    }
}

// ---------------- selective scan: lane = (channel, state) pair -------------------
// wave32: lanes 0..15 = channel c0 states 0..15, lanes 16..31 = channel c1.
__global__ void __launch_bounds__(WG) k_scan(const float* __restrict__ delta,
                                             const float* __restrict__ u,
                                             const float* __restrict__ xp,
                                             const float* __restrict__ A_log,
                                             const float* __restrict__ Bp,
                                             const float* __restrict__ Cp,
                                             const float* __restrict__ Dv,
                                             bf16* __restrict__ yz,
                                             int B, int L, int D, int NC) {
    int lane = threadIdx.x & 31;
    int gwave = (blockIdx.x * WG + threadIdx.x) >> 5;
    int n  = lane & 15;
    int ch = gwave * 2 + (lane >> 4);        // global (b,d) channel index
    int b  = ch / D;
    int d  = ch % D;
    float a  = -__expf(A_log[d * 16 + n]);
    float Bc = Bp[d * 16 + n];
    float Cc = Cp[d * 16 + n];
    float Dd = Dv[d];
    float h = 0.f;
    for (int t = 0; t < L; ++t) {
        size_t idx = (size_t)(b * L + t) * D + d;
        float dt = delta[idx];
        float ut = u[idx];
        float dA = __expf(a * dt);
        h = dA * h + (ut * dt) * Bc;
        float p = h * Cc;
        p += __shfl_xor(p, 1, 32);
        p += __shfl_xor(p, 2, 32);
        p += __shfl_xor(p, 4, 32);
        p += __shfl_xor(p, 8, 32);
        float y = p + Dd * ut;
        float zin = xp[(size_t)(b * L + t) * NC + d];     // first half of xp
        float z = zin * sigmoid_fast(zin);
        if (n == 0) yz[idx] = (bf16)(y * z);
    }
}

extern "C" void kernel_launch(void* const* d_in, const int* in_sizes, int n_in,
                              void* d_out, int out_size, void* d_ws, size_t ws_size,
                              hipStream_t stream) {
    (void)in_sizes; (void)n_in; (void)out_size; (void)ws_size;
    const int B = 2, L = 2048, Dm = 1024, Dn = 2048, N2 = 4096, M = B * L;

    const float* x      = (const float*)d_in[0];
    const float* ln_g   = (const float*)d_in[1];
    const float* ln_b   = (const float*)d_in[2];
    const float* W_in   = (const float*)d_in[3];
    const float* conv_w = (const float*)d_in[4];
    const float* conv_b = (const float*)d_in[5];
    const float* bn_g   = (const float*)d_in[6];
    const float* bn_b   = (const float*)d_in[7];
    const float* A_log  = (const float*)d_in[8];
    const float* Dv     = (const float*)d_in[9];
    const float* Bp     = (const float*)d_in[10];
    const float* Cp     = (const float*)d_in[11];
    const float* W_dt   = (const float*)d_in[12];
    const float* b_dt   = (const float*)d_in[13];
    const float* W_out  = (const float*)d_in[14];
    float* out = (float*)d_out;

    char* w = (char*)d_ws;
    auto alloc = [&](size_t bytes) {
        char* p = w;
        w += (bytes + 255) & ~(size_t)255;
        return p;
    };
    bf16*  xn_bf   = (bf16*) alloc((size_t)M * Dm * 2);
    bf16*  Win_bf  = (bf16*) alloc((size_t)N2 * Dm * 2);
    bf16*  Wdt_bf  = (bf16*) alloc((size_t)Dn * Dn * 2);
    bf16*  Wout_bf = (bf16*) alloc((size_t)Dm * Dn * 2);
    float* xp      = (float*)alloc((size_t)M * N2 * 4);
    float* convu   = (float*)alloc((size_t)M * Dn * 4);   // conv out, then u (f32)
    bf16*  u_bf    = (bf16*) alloc((size_t)M * Dn * 2);   // u bf16, reused as yz bf16
    float* delta   = (float*)alloc((size_t)M * Dn * 4);
    float* sums    = (float*)alloc((size_t)Dn * 4);
    float* sumsq   = (float*)alloc((size_t)Dn * 4);
    float* scale   = (float*)alloc((size_t)Dn * 4);
    float* shift   = (float*)alloc((size_t)Dn * 4);

    // weight conversions (f32 -> bf16)
    k_cvt_bf16<<<4096, WG, 0, stream>>>(W_in, Win_bf, N2 * Dm);
    k_cvt_bf16<<<4096, WG, 0, stream>>>(W_dt, Wdt_bf, Dn * Dn);
    k_cvt_bf16<<<2048, WG, 0, stream>>>(W_out, Wout_bf, Dm * Dn);

    // LayerNorm -> bf16 activations
    k_layernorm<<<M, WG, 0, stream>>>(x, ln_g, ln_b, xn_bf, Dm);

    // GEMM1: xp[M,4096] = xn * W_in^T
    k_gemm_bf16<<<dim3(N2 / 64, M / 128), WG, 0, stream>>>(xn_bf, Win_bf, xp,
                                                           nullptr, nullptr, M, N2, Dm);

    // BN stats must start at zero each call
    (void)hipMemsetAsync(sums,  0, (size_t)Dn * 4, stream);
    (void)hipMemsetAsync(sumsq, 0, (size_t)Dn * 4, stream);

    // depthwise conv on xc = xp[:, 2048:] + stats
    k_dwconv<<<dim3(Dn / WG, B * (L / CONV_TPB)), WG, 0, stream>>>(
        xp, conv_w, conv_b, convu, sums, sumsq, B, L, Dn, N2, Dn);
    k_bn_final<<<(Dn + WG - 1) / WG, WG, 0, stream>>>(sums, sumsq, bn_g, bn_b,
                                                      scale, shift, Dn, 1.0f / (B * L));
    k_bn_silu<<<4096, WG, 0, stream>>>(convu, u_bf, scale, shift, Dn - 1, M * Dn);

    // GEMM2: delta[M,2048] = u * W_dt^T + b_dt
    k_gemm_bf16<<<dim3(Dn / 64, M / 128), WG, 0, stream>>>(u_bf, Wdt_bf, delta,
                                                           b_dt, nullptr, M, Dn, Dn);

    // selective scan; yz (bf16) written into u_bf (free after GEMM2)
    k_scan<<<(B * Dn / 2) / 8, WG, 0, stream>>>(delta, convu, xp, A_log, Bp, Cp, Dv,
                                                u_bf, B, L, Dn, N2);

    // GEMM3: out[M,1024] = yz * W_out^T + x (residual)
    k_gemm_bf16<<<dim3(Dm / 64, M / 128), WG, 0, stream>>>(u_bf, Wout_bf, out,
                                                           nullptr, x, M, Dm, Dn);
}
